// MNISTShareGenerator_72713796321894
// MI455X (gfx1250) — matrix operand order repaired and checked by reading, not compiled, not run on hardware
//
#include <hip/hip_runtime.h>

typedef __attribute__((ext_vector_type(16))) _Float16 v16h;
typedef __attribute__((ext_vector_type(8)))  _Float16 h8;
typedef __attribute__((ext_vector_type(8)))  float    v8f;

__device__ __forceinline__ float lrelu(float v) { return v > 0.0f ? v : 0.2f * v; }

__device__ __forceinline__ v8f wmma_f16(v16h a, v16h b, v8f c) {
  // D(f32) = A(f16 16x32) * B(f16 32x16) + C(f32 16x16)
  return __builtin_amdgcn_wmma_f32_16x16x32_f16(false, a, false, b, (short)0, c, false, false);
}

// CDNA5 async global->LDS copy, 16B per lane (512B per wave32 instruction).
// lds_off: raw LDS byte address (= low 32 bits of the flat shared pointer,
// per ISA 10.2 aperture rule).
__device__ __forceinline__ void async_b128(unsigned lds_off, const _Float16* g) {
  asm volatile("global_load_async_to_lds_b128 %0, %1, off"
               :: "v"(lds_off), "v"((unsigned long long)g)
               : "memory");
}
#define ASYNC_WAIT_6() asm volatile("s_wait_asynccnt 0x6" ::: "memory")
#define ASYNC_WAIT_0() asm volatile("s_wait_asynccnt 0x0" ::: "memory")

// ---------------------------------------------------------------------------
// f32 -> f16 cast (z input and all WMMA-side weights, converted once)
// ---------------------------------------------------------------------------
__global__ void cast_f32_to_f16_kernel(const float* __restrict__ src,
                                       _Float16* __restrict__ dst, int n) {
  int i = blockIdx.x * blockDim.x + threadIdx.x;
  if (i < n) dst[i] = (_Float16)src[i];
}

// ---------------------------------------------------------------------------
// FC layer: out[m,n] = lrelu( sum_k A[m,k] * Wh[n,k] + bias[n] ), stored f16.
// A: f16 [M,K], Wh: f16 [N,K] (pre-converted). Block = 128 thr = 4 waves in a
// 2(M) x 2(N) wave grid; wave tile 32M x 64N; block tile 64M x 128N.
// Double-buffered async-to-LDS pipeline: per k-step the block stages the
// 64x32 A tile (4KB) + 128x32 B tile (8KB) as 24 chunk-instructions
// (6 per wave, 512B each), overlapped with WMMA on the other buffer.
// Grid: (M/64, N/128).
// ---------------------------------------------------------------------------
__global__ void fc_wmma_kernel(const _Float16* __restrict__ A,
                               const _Float16* __restrict__ Wh,
                               const float* __restrict__ bias,
                               _Float16* __restrict__ out,
                               int N, int K) {
  __shared__ alignas(16) _Float16 smem[2 * 6144];  // buf: 2048(A) + 4096(B) halves

  const int lane = threadIdx.x & 31;
  const int wid  = threadIdx.x >> 5;
  const int m0b = blockIdx.x * 64;
  const int n0b = blockIdx.y * 128;
  const int r  = lane & 15;
  const int hi = lane >> 4;
  const int wm = (wid & 1) * 32;
  const int wn = (wid >> 1) * 64;
  const int nk = K >> 5;

  // Stage one 64x32 A + 128x32 B tile into LDS buffer `buf`.
  // Chunk q (0..7 -> A rows q*8..q*8+7; 8..23 -> B cols (q-8)*8..): each chunk
  // is 32 lanes x 16B; wave wid issues chunks [wid*6, wid*6+6).
  auto issue = [&](int k0, int buf) {
    const _Float16* sb = smem + buf * 6144;
#pragma unroll
    for (int i = 0; i < 6; ++i) {
      int q = wid * 6 + i;
      int sub = q < 8 ? q : q - 8;
      int rc  = sub * 8 + (lane >> 2);   // row (A) or col (B)
      int c4  = lane & 3;                // 16B sub-chunk within the 64B row
      if (q < 8) {
        unsigned lds = (unsigned)(size_t)(sb + rc * 32 + c4 * 8);
        async_b128(lds, A + (m0b + rc) * K + k0 + c4 * 8);
      } else {
        unsigned lds = (unsigned)(size_t)(sb + 2048 + rc * 32 + c4 * 8);
        async_b128(lds, Wh + (n0b + rc) * K + k0 + c4 * 8);
      }
    }
  };

  v8f acc[2][4] = {};
  issue(0, 0);

  for (int it = 0; it < nk; ++it) {
    int buf = it & 1;
    if (it + 1 < nk) { issue((it + 1) << 5, buf ^ 1); ASYNC_WAIT_6(); }
    else             { ASYNC_WAIT_0(); }
    __syncthreads();

    const _Float16* As = smem + buf * 6144;
    const _Float16* Bs = As + 2048;

    v16h a[2];
#pragma unroll
    for (int fm = 0; fm < 2; ++fm) {
      const _Float16* p = As + (wm + fm * 16 + r) * 32 + hi * 8;
      h8 lo = *(const h8*)p;
      h8 hh = *(const h8*)(p + 16);
#pragma unroll
      for (int j = 0; j < 8; ++j) { a[fm][j] = lo[j]; a[fm][8 + j] = hh[j]; }
    }
    v16h bfr[4];
#pragma unroll
    for (int fn = 0; fn < 4; ++fn) {
      const _Float16* p = Bs + (wn + fn * 16 + r) * 32 + hi * 8;
      h8 lo = *(const h8*)p;
      h8 hh = *(const h8*)(p + 16);
#pragma unroll
      for (int j = 0; j < 8; ++j) { bfr[fn][j] = lo[j]; bfr[fn][8 + j] = hh[j]; }
    }
#pragma unroll
    for (int fm = 0; fm < 2; ++fm)
#pragma unroll
      for (int fn = 0; fn < 4; ++fn)
        acc[fm][fn] = wmma_f16(a[fm], bfr[fn], acc[fm][fn]);
    __syncthreads();   // protect buf from being overwritten by iteration it+1
  }

  // D layout: M = (lane>>4)*8 + i, N = lane&15
#pragma unroll
  for (int fm = 0; fm < 2; ++fm)
#pragma unroll
    for (int fn = 0; fn < 4; ++fn) {
      int nn = n0b + wn + fn * 16 + r;
      float bv = bias[nn];
#pragma unroll
      for (int i = 0; i < 8; ++i) {
        int mm = m0b + wm + fm * 16 + hi * 8 + i;
        out[mm * N + nn] = (_Float16)lrelu(acc[fm][fn][i] + bv);
      }
    }
}

// ---------------------------------------------------------------------------
// ConvTranspose2d(128->64, k=4, s=2, p=1) + lrelu as implicit GEMM.
// Parity-major M: m = pclass*50176 + b*49 + oyy*7 + oxx   (M = 200704)
// For parity (py,px): taps ky = py+2*tky, kx = px+2*tkx;
// iy = oyy+py+tky-1, ix = oxx+px+tkx-1; K = 128 chans * 4 taps = 512, N = 64.
// Input X f16 [1024,128,7,7]; output Y f16 [1024,64,14,14].
// Block = 4 waves stacked along M (128 rows); grid = 200704/128 = 1568.
// ---------------------------------------------------------------------------
__global__ void convt0_wmma_kernel(const _Float16* __restrict__ X,
                                   const _Float16* __restrict__ W,  // f16 [64,128,4,4]
                                   const float* __restrict__ bias,  // [64]
                                   _Float16* __restrict__ Y) {
  const int lane = threadIdx.x & 31;
  const int wid  = threadIdx.x >> 5;
  const int m0 = blockIdx.x * 128 + wid * 32;
  const int pclass = m0 / 50176;            // uniform per block (50176 % 128 == 0)
  const int py = pclass >> 1, px = pclass & 1;
  const int r  = lane & 15;
  const int hi = lane >> 4;

  v8f acc[2][4] = {};

  for (int k0 = 0; k0 < 512; k0 += 32) {
    v16h a[2];
#pragma unroll
    for (int fm = 0; fm < 2; ++fm) {
      int m = m0 + fm * 16 + r;
      int rem = m - pclass * 50176;
      int bs = rem / 49;
      int pos = rem - bs * 49;
      int oyy = pos / 7, oxx = pos - (pos / 7) * 7;
      const _Float16* base = X + bs * 6272;
#pragma unroll
      for (int j = 0; j < 16; ++j) {
        int k = k0 + hi * 16 + j;
        int ci = k >> 2, t = k & 3;
        int iy = oyy + py + (t >> 1) - 1;
        int ix = oxx + px + (t & 1) - 1;
        a[fm][j] = (iy >= 0 && iy < 7 && ix >= 0 && ix < 7)
                       ? base[ci * 49 + iy * 7 + ix] : (_Float16)0.0f;
      }
    }
    v16h bf[4];
#pragma unroll
    for (int fn = 0; fn < 4; ++fn) {
      int o = fn * 16 + r;
#pragma unroll
      for (int j = 0; j < 16; ++j) {
        int k = k0 + hi * 16 + j;
        int ci = k >> 2, t = k & 3;
        int ky = py + 2 * (t >> 1);
        int kx = px + 2 * (t & 1);
        bf[fn][j] = W[o * 2048 + ci * 16 + ky * 4 + kx];
      }
    }
#pragma unroll
    for (int fm = 0; fm < 2; ++fm)
#pragma unroll
      for (int fn = 0; fn < 4; ++fn)
        acc[fm][fn] = wmma_f16(a[fm], bf[fn], acc[fm][fn]);
  }

#pragma unroll
  for (int fm = 0; fm < 2; ++fm)
#pragma unroll
    for (int i = 0; i < 8; ++i) {
      int m = m0 + fm * 16 + hi * 8 + i;
      int rem = m - pclass * 50176;
      int bs = rem / 49;
      int pos = rem - bs * 49;
      int oyy = pos / 7, oxx = pos - (pos / 7) * 7;
      int oy = 2 * oyy + py, ox = 2 * oxx + px;
#pragma unroll
      for (int fn = 0; fn < 4; ++fn) {
        int o = fn * 16 + r;
        Y[bs * 12544 + o * 196 + oy * 14 + ox] =
            (_Float16)lrelu(acc[fm][fn][i] + bias[o]);
      }
    }
}

// ---------------------------------------------------------------------------
// Per-sample branch ConvTranspose2d(64->16, k=4, s=2, p=1) + lrelu.
// Weights selected by g_idx[b]. Sample-major M with parity segments padded
// to 224 rows: m = b*896 + pclass*224 + pos (pos<196 valid). M = 917504.
// K = 64*4 = 256, N = 16. Block = 4 waves along M (896 % 128 == 0 so b is
// block-uniform; 224 % 32 == 0 so parity is wave-uniform). Grid = 7168.
// ---------------------------------------------------------------------------
__global__ void convta_wmma_kernel(const _Float16* __restrict__ X,   // lat [1024,64,14,14]
                                   const _Float16* __restrict__ WA,  // f16 [10,16,64,4,4]
                                   const float* __restrict__ BA,     // [10,16]
                                   const int* __restrict__ g_idx,
                                   _Float16* __restrict__ Y) {       // [1024,16,28,28]
  const int lane = threadIdx.x & 31;
  const int wid  = threadIdx.x >> 5;
  const int m0 = blockIdx.x * 128 + wid * 32;
  const int bs = m0 / 896;
  const int g  = g_idx[bs];
  const int pclass = (m0 - bs * 896) / 224;
  const int py = pclass >> 1, px = pclass & 1;
  const int r  = lane & 15;
  const int hi = lane >> 4;
  const _Float16* base = X + bs * 12544;

  v8f acc[2] = {};

  for (int k0 = 0; k0 < 256; k0 += 32) {
    v16h a[2];
#pragma unroll
    for (int fm = 0; fm < 2; ++fm) {
      int m = m0 + fm * 16 + r;
      int pos = (m - bs * 896) - pclass * 224;
      bool valid = pos < 196;
      int oyy = pos / 14, oxx = pos - (pos / 14) * 14;
#pragma unroll
      for (int j = 0; j < 16; ++j) {
        int k = k0 + hi * 16 + j;
        int ci = k >> 2, t = k & 3;
        int iy = oyy + py + (t >> 1) - 1;
        int ix = oxx + px + (t & 1) - 1;
        a[fm][j] = (valid && iy >= 0 && iy < 14 && ix >= 0 && ix < 14)
                       ? base[ci * 196 + iy * 14 + ix] : (_Float16)0.0f;
      }
    }
    v16h bf;
#pragma unroll
    for (int j = 0; j < 16; ++j) {
      int k = k0 + hi * 16 + j;
      int ci = k >> 2, t = k & 3;
      int ky = py + 2 * (t >> 1);
      int kx = px + 2 * (t & 1);
      bf[j] = WA[(g * 16 + r) * 1024 + ci * 16 + ky * 4 + kx];
    }
    acc[0] = wmma_f16(a[0], bf, acc[0]);
    acc[1] = wmma_f16(a[1], bf, acc[1]);
  }

  float bv = BA[g * 16 + r];
#pragma unroll
  for (int fm = 0; fm < 2; ++fm)
#pragma unroll
    for (int i = 0; i < 8; ++i) {
      int m = m0 + fm * 16 + hi * 8 + i;
      int pos = (m - bs * 896) - pclass * 224;
      if (pos < 196) {
        int oyy = pos / 14, oxx = pos - (pos / 14) * 14;
        int oy = 2 * oyy + py, ox = 2 * oxx + px;
        Y[bs * 12544 + r * 784 + oy * 28 + ox] =
            (_Float16)lrelu(acc[fm][i] + bv);
      }
    }
}

// ---------------------------------------------------------------------------
// Final ConvTranspose2d(16->1, k=3, s=1, p=1) == 3x3 conv pad 1, + tanh.
// Only 0.23 GF with Cout=1 -> VALU. Output f32 [1024,1,28,28].
// ---------------------------------------------------------------------------
__global__ void convb_tanh_kernel(const _Float16* __restrict__ YA,  // [1024,16,28,28]
                                  const float* __restrict__ WB,     // [10,1,16,3,3]
                                  const float* __restrict__ BB,     // [10,1]
                                  const int* __restrict__ g_idx,
                                  float* __restrict__ out) {
  int idx = blockIdx.x * blockDim.x + threadIdx.x;
  if (idx >= 1024 * 784) return;
  int bs = idx / 784;
  int pos = idx - bs * 784;
  int oy = pos / 28, ox = pos - (pos / 28) * 28;
  int g = g_idx[bs];
  const _Float16* base = YA + bs * 12544;
  const float* w = WB + g * 144;
  float acc = BB[g];
  for (int c = 0; c < 16; ++c) {
#pragma unroll
    for (int ky = 0; ky < 3; ++ky) {
      int iy = oy + ky - 1;
      if (iy < 0 || iy >= 28) continue;
#pragma unroll
      for (int kx = 0; kx < 3; ++kx) {
        int ix = ox + kx - 1;
        if (ix < 0 || ix >= 28) continue;
        acc += (float)base[c * 784 + iy * 28 + ix] * w[c * 9 + ky * 3 + kx];
      }
    }
  }
  out[idx] = tanhf(acc);
}

// ---------------------------------------------------------------------------
extern "C" void kernel_launch(void* const* d_in, const int* in_sizes, int n_in,
                              void* d_out, int out_size, void* d_ws, size_t ws_size,
                              hipStream_t stream) {
  const float* z   = (const float*)d_in[0];
  const int* g_idx = (const int*)d_in[1];
  const float* W1  = (const float*)d_in[2];
  const float* b1  = (const float*)d_in[3];
  const float* W2  = (const float*)d_in[4];
  const float* b2  = (const float*)d_in[5];
  const float* w0  = (const float*)d_in[6];
  const float* b0  = (const float*)d_in[7];
  const float* wa  = (const float*)d_in[8];
  const float* ba  = (const float*)d_in[9];
  const float* wb  = (const float*)d_in[10];
  const float* bb  = (const float*)d_in[11];
  float* out = (float*)d_out;

  // Workspace layout (f16 activations + f16 weight caches, ~76.6 MB)
  char* ws = (char*)d_ws;
  _Float16* zh  = (_Float16*)(ws + 0);          // [1024,128]       256 KB
  _Float16* h1  = (_Float16*)(ws + 262144);     // [1024,1024]      2 MB
  _Float16* h2  = (_Float16*)(ws + 2359296);    // [1024,6272]      12.25 MB
  _Float16* lat = (_Float16*)(ws + 15204352);   // [1024,64,14,14]  24.5 MB
  _Float16* ya  = (_Float16*)(ws + 40894464);   // [1024,16,28,28]  24.5 MB
  _Float16* W1h = (_Float16*)(ws + 66584576);   // [1024,128]       256 KB
  _Float16* W2h = (_Float16*)(ws + 66846720);   // [6272,1024]      12.25 MB
  _Float16* w0h = (_Float16*)(ws + 79691776);   // [64,128,4,4]     256 KB
  _Float16* wah = (_Float16*)(ws + 79953920);   // [10,16,64,4,4]   320 KB

  // One-time f16 conversions (z + all WMMA-side weights)
  cast_f32_to_f16_kernel<<<512,   256, 0, stream>>>(z,  zh,  1024 * 128);
  cast_f32_to_f16_kernel<<<512,   256, 0, stream>>>(W1, W1h, 1024 * 128);
  cast_f32_to_f16_kernel<<<25088, 256, 0, stream>>>(W2, W2h, 6272 * 1024);
  cast_f32_to_f16_kernel<<<512,   256, 0, stream>>>(w0, w0h, 64 * 128 * 16);
  cast_f32_to_f16_kernel<<<640,   256, 0, stream>>>(wa, wah, 10 * 16 * 64 * 16);

  // FC1: M=1024 N=1024 K=128
  fc_wmma_kernel<<<dim3(16, 8), 128, 0, stream>>>(zh, W1h, b1, h1, 1024, 128);
  // FC2: M=1024 N=6272 K=1024
  fc_wmma_kernel<<<dim3(16, 49), 128, 0, stream>>>(h1, W2h, b2, h2, 6272, 1024);
  // ConvT0 implicit GEMM: M=200704 N=64 K=512
  convt0_wmma_kernel<<<1568, 128, 0, stream>>>(h2, w0h, b0, lat);
  // Branch ConvT implicit GEMM: M=917504 N=16 K=256
  convta_wmma_kernel<<<7168, 128, 0, stream>>>(lat, wah, ba, g_idx, ya);
  // Final 3x3 conv + tanh
  convb_tanh_kernel<<<3136, 256, 0, stream>>>(ya, wb, bb, g_idx, out);

  (void)in_sizes; (void)n_in; (void)out_size; (void)ws_size;
}
